// MultiHeadAttention_21552145891755
// MI455X (gfx1250) — matrix operand order repaired and checked by reading, not compiled, not run on hardware
//
#include <hip/hip_runtime.h>
#include <hip/hip_bf16.h>

// ---------- types ----------
typedef __attribute__((ext_vector_type(16))) __bf16 v16bf;
typedef __attribute__((ext_vector_type(8)))  __bf16 v8bf;
typedef __attribute__((ext_vector_type(8)))  float  v8f;

#define N_EMBD 384
#define N_HEAD 6
#define HEAD_DIM 64
#define SEQ_T 2048
#define BATCH 8
#define M_TOT (BATCH * SEQ_T)   // 16384

#define AS1 __attribute__((address_space(1)))
#define AS3 __attribute__((address_space(3)))

// CDNA5 data-movement path selection (probe-guarded)
#if __has_builtin(__builtin_amdgcn_tensor_load_to_lds) && __has_builtin(__builtin_amdgcn_s_wait_tensorcnt)
#define HAVE_TDM 1
#elif __has_builtin(__builtin_amdgcn_global_load_async_to_lds_b128) && __has_builtin(__builtin_amdgcn_s_wait_asynccnt)
#define HAVE_ASYNC 1
#endif

// ---------- helpers ----------
__device__ __forceinline__ __bf16 f2bf(float f) {
    union { float f; unsigned u; } x; x.f = f;
    unsigned r = x.u + 0x7FFFu + ((x.u >> 16) & 1u);   // round-to-nearest-even
    unsigned short h = (unsigned short)(r >> 16);
    return __builtin_bit_cast(__bf16, h);
}

__device__ __forceinline__ v8bf ld8(const __bf16* p) { return *(const v8bf*)p; }

__device__ __forceinline__ v16bf mk16(v8bf a, v8bf b) {
    v16bf r;
#pragma unroll
    for (int i = 0; i < 8; ++i) { r[i] = a[i]; r[i + 8] = b[i]; }
    return r;
}
__device__ __forceinline__ v16bf ld16(const __bf16* p) { return mk16(ld8(p), ld8(p + 8)); }

__device__ __forceinline__ v8f wmma_bf16(v16bf a, v16bf b, v8f c) {
    return __builtin_amdgcn_wmma_f32_16x16x32_bf16(false, a, false, b, (short)0, c, false, false);
}

__device__ __forceinline__ float redmax16(float v) {
#pragma unroll
    for (int m = 1; m < 16; m <<= 1) v = fmaxf(v, __shfl_xor(v, m, 32));
    return v;
}
__device__ __forceinline__ float redsum16(float v) {
#pragma unroll
    for (int m = 1; m < 16; m <<= 1) v += __shfl_xor(v, m, 32);
    return v;
}

__device__ __forceinline__ unsigned lds_off(void* p) {
    return (unsigned)(unsigned long long)(AS3 void*)p;
}

#if defined(HAVE_TDM)
typedef __attribute__((ext_vector_type(4))) unsigned ui32x4;
typedef __attribute__((ext_vector_type(8))) int i32x8;
typedef __attribute__((ext_vector_type(4))) int i32x4;

// 2D tile DMA: global(row-major, stride0 elems) -> LDS with per-row padding.
// dims/strides in elements (data_size = 2 bytes). pad codes per D# tables.
__device__ __forceinline__ void tdm_load_2d(const void* gaddr, unsigned lds_addr,
                                            unsigned tdim0, unsigned tdim1, unsigned stride0,
                                            unsigned tile0, unsigned tile1,
                                            unsigned pad_int_code, unsigned pad_amt_code)
{
    unsigned long long ga = (unsigned long long)gaddr;
    ui32x4 g0;
    g0[0] = 1u;                                                   // count=1 (valid)
    g0[1] = lds_addr;                                             // lds_addr
    g0[2] = (unsigned)ga;                                         // global_addr[31:0]
    g0[3] = ((unsigned)(ga >> 32) & 0x01FFFFFFu) | (2u << 30);    // addr[56:32], type=2
    i32x8 g1;
    g1[0] = (int)((1u << 16) | (1u << 20) |                       // data_size=2B, pad_enable
                  (pad_int_code << 22) | (pad_amt_code << 25));
    g1[1] = (int)(tdim0 << 16);                                   // tensor_dim0[15:0]
    g1[2] = (int)((tdim0 >> 16) | (tdim1 << 16));                 // tdim0 hi | tensor_dim1 lo
    g1[3] = (int)((tdim1 >> 16) | (tile0 << 16));                 // tdim1 hi | tile_dim0
    g1[4] = (int)(tile1 & 0xFFFFu);                               // tile_dim1 (tile_dim2=0)
    g1[5] = (int)stride0;                                         // tensor_dim0_stride lo32
    g1[6] = 0;
    g1[7] = 0;
    i32x4 z4 = {0, 0, 0, 0};
#if __clang_major__ >= 23
    i32x8 z8 = {0, 0, 0, 0, 0, 0, 0, 0};
    __builtin_amdgcn_tensor_load_to_lds(g0, g1, z4, z4, z8, 0);
#else
    __builtin_amdgcn_tensor_load_to_lds(g0, g1, z4, z4, 0);
#endif
}
#endif

// ---------- f32 -> bf16 conversion ----------
__global__ void cvt_bf16_kernel(const float* __restrict__ src, __bf16* __restrict__ dst, int n) {
    int i = blockIdx.x * blockDim.x + threadIdx.x;
    int stride = gridDim.x * blockDim.x;
    for (; i < n; i += stride) dst[i] = f2bf(src[i]);
}

// ---------- fused QKV GEMM + RoPE + RMSNorm ----------
// grid (128, 6, 3), block 256. Each wave: 16 tokens x one head (64 dims).
// q/k -> bf16 [B,H,T,D];  v -> bf16 [B,H,D,T] (pre-transposed for TDM staging).
__global__ __launch_bounds__(256) void qkv_kernel(
    const __bf16* __restrict__ xb,
    const __bf16* __restrict__ Wqb, const __bf16* __restrict__ Wkb, const __bf16* __restrict__ Wvb,
    const float* __restrict__ cosp, const float* __restrict__ sinp,
    __bf16* __restrict__ qb, __bf16* __restrict__ kb, __bf16* __restrict__ vb)
{
    const int wid  = threadIdx.x >> 5;
    const int lane = threadIdx.x & 31;
    const int ln = lane & 15, sel = lane >> 4;
    const int m0 = (blockIdx.x * 8 + wid) * 16;
    const int h  = blockIdx.y;
    const int which = blockIdx.z;
    const __bf16* W = (which == 0) ? Wqb : (which == 1) ? Wkb : Wvb;
    const int n0 = h * HEAD_DIM;

    v8f acc[4] = {};
#pragma unroll 4
    for (int kk = 0; kk < N_EMBD / 32; ++kk) {
        const int k0 = kk * 32;
        const __bf16* ar = xb + (size_t)(m0 + ln) * N_EMBD + k0 + sel * 8;
        v16bf a = mk16(ld8(ar), ld8(ar + 16));
#pragma unroll
        for (int t = 0; t < 4; ++t) {
            const __bf16* br = W + (size_t)(n0 + t * 16 + ln) * N_EMBD + k0 + sel * 8;
            v16bf b = mk16(ld8(br), ld8(br + 16));
            acc[t] = wmma_bf16(a, b, acc[t]);
        }
    }

    const int ro = sel * 8;
    if (which == 2) {
        // V: convert + store transposed [B,H,D,T]
#pragma unroll
        for (int r = 0; r < 8; ++r) {
            int m = m0 + r + ro;
            int bB = m >> 11, tt = m & (SEQ_T - 1);
            __bf16* dst = vb + (size_t)((bB * N_HEAD + h) * HEAD_DIM) * SEQ_T;
#pragma unroll
            for (int t = 0; t < 4; ++t)
                dst[(size_t)(t * 16 + ln) * SEQ_T + tt] = f2bf(acc[t][r]);
        }
    } else {
        __bf16* base = (which == 0) ? qb : kb;
#pragma unroll
        for (int r = 0; r < 8; ++r) {
            int m = m0 + r + ro;
            int bB = m >> 11, tt = m & (SEQ_T - 1);
            // RoPE: pair C-tiles (0,2) and (1,3): x1 = dims 0..31, x2 = dims 32..63
            float c0 = cosp[tt * 32 + ln],      s0 = sinp[tt * 32 + ln];
            float c1 = cosp[tt * 32 + 16 + ln], s1 = sinp[tt * 32 + 16 + ln];
            float x1a = acc[0][r], x1b = acc[1][r], x2a = acc[2][r], x2b = acc[3][r];
            float y0 =  x1a * c0 + x2a * s0;
            float y1 =  x1b * c1 + x2b * s1;
            float y2 =  x2a * c0 - x1a * s0;
            float y3 =  x2b * c1 - x1b * s1;
            // RMSNorm over 64 dims of this row (4 elems/lane x 16 lanes)
            float ss = y0 * y0 + y1 * y1 + y2 * y2 + y3 * y3;
            ss = redsum16(ss);
            float sc = rsqrtf(ss * (1.0f / 64.0f) + 1e-5f);
            __bf16* dst = base + (size_t)((bB * N_HEAD + h) * SEQ_T + tt) * HEAD_DIM;
            dst[ln]      = f2bf(y0 * sc);
            dst[16 + ln] = f2bf(y1 * sc);
            dst[32 + ln] = f2bf(y2 * sc);
            dst[48 + ln] = f2bf(y3 * sc);
        }
    }
}

// ---------- flash attention ----------
// grid (16, 6, 8) = (T/128, H, B), block 256 (8 waves, 16 query rows each).
// K/V blocks double-buffered in LDS, staged by the Tensor Data Mover.
#define KS_STRIDE 72   // 64 + 8 bf16 pad (TDM pad: 128B rows + 16B)
#define VT_STRIDE 40   // 32 + 8 bf16 pad (TDM pad:  64B rows + 16B)

__device__ __forceinline__ void sync_stage(const __bf16* kB, const __bf16* vB, int k0r,
                                           __bf16* ksb, __bf16* vtb, int tid) {
    int krow = tid >> 3, kc = (tid & 7) * 8;
    uint4 kd = *(const uint4*)(kB + (size_t)(k0r + krow) * HEAD_DIM + kc);
    *(uint4*)(&ksb[krow * KS_STRIDE + kc]) = kd;
    int vrow = tid >> 2, vc = (tid & 3) * 8;
    uint4 vd = *(const uint4*)(vB + (size_t)vrow * SEQ_T + k0r + vc);
    *(uint4*)(&vtb[vrow * VT_STRIDE + vc]) = vd;
}

#if defined(HAVE_ASYNC)
__device__ __forceinline__ void async_stage(const __bf16* kB, const __bf16* vB, int k0r,
                                            __bf16* ksb, __bf16* vtb, int tid) {
    int krow = tid >> 3, kc = (tid & 7) * 8;
    __builtin_amdgcn_global_load_async_to_lds_b128(
        (const AS1 void*)(kB + (size_t)(k0r + krow) * HEAD_DIM + kc),
        (AS3 void*)&ksb[krow * KS_STRIDE + kc], 0, 0);
    int vrow = tid >> 2, vc = (tid & 3) * 8;
    __builtin_amdgcn_global_load_async_to_lds_b128(
        (const AS1 void*)(vB + (size_t)vrow * SEQ_T + k0r + vc),
        (AS3 void*)&vtb[vrow * VT_STRIDE + vc], 0, 0);
}
#endif

__global__ __launch_bounds__(256) void attn_kernel(
    const __bf16* __restrict__ qb, const __bf16* __restrict__ kb, const __bf16* __restrict__ vb,
    __bf16* __restrict__ yb)
{
    __shared__ __bf16 Ks[2][32 * KS_STRIDE];   // K block, row-major [key][d], padded
    __shared__ __bf16 Vt[2][64 * VT_STRIDE];   // V block, [d][key], padded
    __shared__ __bf16 Ps[8 * 16 * 32];         // per-wave P staging [row][key]

    const int tid  = threadIdx.x;
    const int wid  = tid >> 5;
    const int lane = tid & 31;
    const int ln = lane & 15, sel = lane >> 4;
    const int h = blockIdx.y, bB = blockIdx.z;
    const int t0 = blockIdx.x * 128 + wid * 16;
    const size_t base = (size_t)((bB * N_HEAD + h) * SEQ_T) * HEAD_DIM;
    const __bf16* qB = qb + base;
    const __bf16* kB = kb + base;
    const __bf16* vB = vb + base;   // [D][T] tile for this (b,h)

    // Q A-fragments, loaded once (d chunks 0..31 and 32..63)
    const __bf16* qrow = qB + (size_t)(t0 + ln) * HEAD_DIM + sel * 8;
    v16bf qa0 = mk16(ld8(qrow),      ld8(qrow + 16));
    v16bf qa1 = mk16(ld8(qrow + 32), ld8(qrow + 48));

    float m_[8], l_[8];
    v8f o[4] = {};
#pragma unroll
    for (int r = 0; r < 8; ++r) { m_[r] = -1e30f; l_[r] = 0.0f; }

    __bf16* pw = Ps + wid * 512;
    const int NB = SEQ_T / 32;

#if defined(HAVE_TDM)
    unsigned ksOff[2] = { lds_off(&Ks[0][0]), lds_off(&Ks[1][0]) };
    unsigned vtOff[2] = { lds_off(&Vt[0][0]), lds_off(&Vt[1][0]) };
    if (wid == 0) {
        tdm_load_2d(kB, ksOff[0], 64, 32,   64, 64, 32, 4, 3);
        tdm_load_2d(vB, vtOff[0], 32, 64, 2048, 32, 64, 3, 3);
        __builtin_amdgcn_s_wait_tensorcnt(0);
    }
#elif defined(HAVE_ASYNC)
    async_stage(kB, vB, 0, Ks[0], Vt[0], tid);
    __builtin_amdgcn_s_wait_asynccnt(0);
#else
    sync_stage(kB, vB, 0, Ks[0], Vt[0], tid);
#endif
    __syncthreads();

    for (int jb = 0; jb < NB; ++jb) {
        const int cur = jb & 1, nxt = cur ^ 1;
        const bool have_next = (jb + 1 < NB);
        const int k1 = (jb + 1) * 32;

        // kick off staging of the next block before computing on this one
#if defined(HAVE_TDM)
        if (wid == 0 && have_next) {
            tdm_load_2d(kB + (size_t)k1 * HEAD_DIM, ksOff[nxt], 64, 32,   64, 64, 32, 4, 3);
            tdm_load_2d(vB + k1,                    vtOff[nxt], 32, 64, 2048, 32, 64, 3, 3);
        }
#elif defined(HAVE_ASYNC)
        if (have_next) async_stage(kB, vB, k1, Ks[nxt], Vt[nxt], tid);
#endif

        const __bf16* ks = Ks[cur];
        const __bf16* vt = Vt[cur];

        // S = Q K^T : 2 key tiles x 2 K-dim chunks
        v8f s0 = {}, s1 = {};
        {
            v16bf b00 = mk16(ld8(&ks[(ln)      * KS_STRIDE + sel * 8]), ld8(&ks[(ln)      * KS_STRIDE + sel * 8 + 16]));
            v16bf b01 = mk16(ld8(&ks[(ln)      * KS_STRIDE + 32 + sel * 8]), ld8(&ks[(ln)      * KS_STRIDE + 48 + sel * 8]));
            s0 = wmma_bf16(qa0, b00, s0);
            s0 = wmma_bf16(qa1, b01, s0);
            v16bf b10 = mk16(ld8(&ks[(16 + ln) * KS_STRIDE + sel * 8]), ld8(&ks[(16 + ln) * KS_STRIDE + sel * 8 + 16]));
            v16bf b11 = mk16(ld8(&ks[(16 + ln) * KS_STRIDE + 32 + sel * 8]), ld8(&ks[(16 + ln) * KS_STRIDE + 48 + sel * 8]));
            s1 = wmma_bf16(qa0, b10, s1);
            s1 = wmma_bf16(qa1, b11, s1);
        }

        // online softmax per C-row (VGPR r, lane half)
#pragma unroll
        for (int r = 0; r < 8; ++r) {
            float a = s0[r] * 0.125f;   // 1/sqrt(64)
            float b = s1[r] * 0.125f;
            float mx = redmax16(fmaxf(a, b));
            float mn = fmaxf(m_[r], mx);
            float pa = __expf(a - mn);
            float pb = __expf(b - mn);
            float alpha = __expf(m_[r] - mn);
            float rs = redsum16(pa + pb);
            l_[r] = l_[r] * alpha + rs;
            m_[r] = mn;
#pragma unroll
            for (int t = 0; t < 4; ++t) o[t][r] *= alpha;
            int row = r + sel * 8;
            pw[row * 32 + ln]      = f2bf(pa);
            pw[row * 32 + 16 + ln] = f2bf(pb);
        }

        // reload P in A-fragment layout, accumulate O += P V
        const __bf16* pr = pw + ln * 32 + sel * 8;
        v16bf pfa = mk16(ld8(pr), ld8(pr + 16));
#pragma unroll
        for (int t = 0; t < 4; ++t) {
            v16bf vf = mk16(ld8(&vt[(t * 16 + ln) * VT_STRIDE + sel * 8]), ld8(&vt[(t * 16 + ln) * VT_STRIDE + sel * 8 + 16]));
            o[t] = wmma_bf16(pfa, vf, o[t]);
        }

        // finish staging of the next block, then block-wide handoff
#if defined(HAVE_TDM)
        if (wid == 0 && have_next) __builtin_amdgcn_s_wait_tensorcnt(0);
#elif defined(HAVE_ASYNC)
        if (have_next) __builtin_amdgcn_s_wait_asynccnt(0);
#else
        if (have_next) sync_stage(kB, vB, k1, Ks[nxt], Vt[nxt], tid);
#endif
        __syncthreads();
    }

    // epilogue: normalize and store as bf16 [B, T, C]
#pragma unroll
    for (int r = 0; r < 8; ++r) {
        float inv = 1.0f / l_[r];
        int tt = t0 + r + sel * 8;
        __bf16* dst = yb + (size_t)(bB * SEQ_T + tt) * N_EMBD + h * HEAD_DIM;
#pragma unroll
        for (int t = 0; t < 4; ++t) dst[t * 16 + ln] = f2bf(o[t][r] * inv);
    }
}

// ---------- output projection: out = y @ Wo^T (f32 output) ----------
// grid (128, 6), block 256.
__global__ __launch_bounds__(256) void proj_kernel(
    const __bf16* __restrict__ yb, const __bf16* __restrict__ Wob, float* __restrict__ out)
{
    const int wid  = threadIdx.x >> 5;
    const int lane = threadIdx.x & 31;
    const int ln = lane & 15, sel = lane >> 4;
    const int m0 = (blockIdx.x * 8 + wid) * 16;
    const int n0 = blockIdx.y * 64;

    v8f acc[4] = {};
#pragma unroll 4
    for (int kk = 0; kk < N_EMBD / 32; ++kk) {
        const int k0 = kk * 32;
        const __bf16* ar = yb + (size_t)(m0 + ln) * N_EMBD + k0 + sel * 8;
        v16bf a = mk16(ld8(ar), ld8(ar + 16));
#pragma unroll
        for (int t = 0; t < 4; ++t) {
            const __bf16* br = Wob + (size_t)(n0 + t * 16 + ln) * N_EMBD + k0 + sel * 8;
            v16bf b = mk16(ld8(br), ld8(br + 16));
            acc[t] = wmma_bf16(a, b, acc[t]);
        }
    }
#pragma unroll
    for (int r = 0; r < 8; ++r) {
        int m = m0 + r + sel * 8;
#pragma unroll
        for (int t = 0; t < 4; ++t)
            out[(size_t)m * N_EMBD + n0 + t * 16 + ln] = acc[t][r];
    }
}

// ---------- host launch ----------
extern "C" void kernel_launch(void* const* d_in, const int* in_sizes, int n_in,
                              void* d_out, int out_size, void* d_ws, size_t ws_size,
                              hipStream_t stream) {
    const float* x    = (const float*)d_in[0];
    const float* Wq   = (const float*)d_in[1];
    const float* Wk   = (const float*)d_in[2];
    const float* Wv   = (const float*)d_in[3];
    const float* Wo   = (const float*)d_in[4];
    const float* cosp = (const float*)d_in[5];
    const float* sinp = (const float*)d_in[6];
    float* out = (float*)d_out;

    // workspace layout (256B aligned)
    char* ws = (char*)d_ws;
    size_t off = 0;
    auto take = [&](size_t bytes) { char* p = ws + off; off = (off + bytes + 255) & ~(size_t)255; return p; };
    const size_t xbytes = (size_t)M_TOT * N_EMBD * 2;
    const size_t wbytes = (size_t)N_EMBD * N_EMBD * 2;
    const size_t hbytes = (size_t)BATCH * N_HEAD * SEQ_T * HEAD_DIM * 2;
    __bf16* xb  = (__bf16*)take(xbytes);
    __bf16* Wqb = (__bf16*)take(wbytes);
    __bf16* Wkb = (__bf16*)take(wbytes);
    __bf16* Wvb = (__bf16*)take(wbytes);
    __bf16* Wob = (__bf16*)take(wbytes);
    __bf16* qb  = (__bf16*)take(hbytes);
    __bf16* kb  = (__bf16*)take(hbytes);
    __bf16* vb  = (__bf16*)take(hbytes);   // [B,H,D,T]
    __bf16* yb  = (__bf16*)take(xbytes);
    (void)ws_size; (void)in_sizes; (void)n_in; (void)out_size;

    const int nX = M_TOT * N_EMBD;
    const int nW = N_EMBD * N_EMBD;
    cvt_bf16_kernel<<<2048, 256, 0, stream>>>(x,  xb,  nX);
    cvt_bf16_kernel<<<144,  256, 0, stream>>>(Wq, Wqb, nW);
    cvt_bf16_kernel<<<144,  256, 0, stream>>>(Wk, Wkb, nW);
    cvt_bf16_kernel<<<144,  256, 0, stream>>>(Wv, Wvb, nW);
    cvt_bf16_kernel<<<144,  256, 0, stream>>>(Wo, Wob, nW);

    dim3 gq(M_TOT / 16 / 8, N_HEAD, 3);
    qkv_kernel<<<gq, 256, 0, stream>>>(xb, Wqb, Wkb, Wvb, cosp, sinp, qb, kb, vb);

    dim3 ga(SEQ_T / 128, N_HEAD, BATCH);
    attn_kernel<<<ga, 256, 0, stream>>>(qb, kb, vb, yb);

    dim3 gp(M_TOT / 16 / 8, N_EMBD / 64, 1);
    proj_kernel<<<gp, 256, 0, stream>>>(yb, Wob, out);
}
